// _LSTMClassifier_52295521796772
// MI455X (gfx1250) — matrix operand order repaired and checked by reading, not compiled, not run on hardware
//
#include <hip/hip_runtime.h>
#include <hip/hip_bf16.h>

#define UNITT   20
#define HID     36
#define IN_DIM  80
#define GATES   144
#define BT      16     // batch rows per block (one WMMA N tile)
#define NWAVES  9      // one 16-gate column tile per wave
#define THREADS (NWAVES * 32)

typedef __attribute__((ext_vector_type(16))) __bf16 v16bf;
typedef __attribute__((ext_vector_type(8)))  float  v8f;

union BF16x16 { v16bf v; unsigned short u[16]; uint4 q[2]; };
union F32x8   { v8f   v; float f[8]; };

// fp32 -> bf16 round-to-nearest-even (bitwise, avoids scalar __bf16 arithmetic)
__device__ __forceinline__ unsigned short f2bf(float f) {
    unsigned int u = __float_as_uint(f);
    return (unsigned short)((u + 0x7FFFu + ((u >> 16) & 1u)) >> 16);
}
__device__ __forceinline__ float sigm(float x) {
    return __builtin_amdgcn_rcpf(1.0f + __expf(-x));
}
__device__ __forceinline__ float tanh_f(float x) {
    return 2.0f * __builtin_amdgcn_rcpf(1.0f + __expf(-2.0f * x)) - 1.0f;
}

__global__ __launch_bounds__(THREADS, 1)
void lstm_fused_kernel(const float* __restrict__ x,
                       const float* __restrict__ h0,
                       const float* __restrict__ c0,
                       const float* __restrict__ W_ih,
                       const float* __restrict__ W_hh,
                       const float* __restrict__ b_ih,
                       const float* __restrict__ b_hh,
                       const float* __restrict__ fc1_w,
                       const float* __restrict__ fc1_b,
                       const float* __restrict__ fc2_w,
                       const float* __restrict__ fc2_b,
                       float* __restrict__ out,
                       int T, int S)
{
    // LDS: x tile (bf16, K padded to 96), h state (bf16, K padded to 64),
    // gate exchange buffer (fp32), fc1 activations.
    __shared__ alignas(16) unsigned short xs[BT][96];
    __shared__ alignas(16) unsigned short hs[BT][64];
    __shared__ alignas(16) float          gbuf[BT][GATES];
    __shared__ alignas(16) float          f1[BT][16];

    const int tid   = threadIdx.x;
    const int wave  = tid >> 5;          // 0..8, owns gates [16*wave, 16*wave+16)
    const int lane  = tid & 31;
    const int lhalf = lane >> 4;         // 0 or 1
    const int lmod  = lane & 15;
    const int b0    = blockIdx.x * BT;

    // ---- one-time: weight A-tiles in registers (bf16) ------------------
    // A 16x32 bf16 layout: M = lane&15, K = 32*kb + e + 8*((e>>3) + lane>>4)
    BF16x16 a_ih[3], a_hh[2];
    {
        const int gate = 16 * wave + lmod;
        #pragma unroll
        for (int kb = 0; kb < 3; ++kb) {
            #pragma unroll
            for (int e = 0; e < 16; ++e) {
                int K = 32 * kb + e + 8 * ((e >> 3) + lhalf);
                float v = (K < IN_DIM) ? W_ih[gate * IN_DIM + K] : 0.0f;
                a_ih[kb].u[e] = f2bf(v);
            }
        }
        #pragma unroll
        for (int kb = 0; kb < 2; ++kb) {
            #pragma unroll
            for (int e = 0; e < 16; ++e) {
                int K = 32 * kb + e + 8 * ((e >> 3) + lhalf);
                float v = (K < HID) ? W_hh[gate * HID + K] : 0.0f;
                a_hh[kb].u[e] = f2bf(v);
            }
        }
    }

    // ---- bias folded into C/D lane layout: gate = 16w + i + 8*lhalf ----
    F32x8 biasv;
    #pragma unroll
    for (int i = 0; i < 8; ++i) {
        int gate = 16 * wave + 8 * lhalf + i;
        biasv.f[i] = b_ih[gate] + b_hh[gate];
    }

    // ---- per-lane c-state: 2 (batch,hid) pairs, wave owns hid [4w,4w+4)
    float cst[2];
    int   bq[2], hq[2];
    #pragma unroll
    for (int q = 0; q < 2; ++q) {
        int idx = lane + 32 * q;
        bq[q]  = idx & 15;
        hq[q]  = 4 * wave + (idx >> 4);
        cst[q] = c0[(size_t)(b0 + bq[q]) * HID + hq[q]];
    }

    // ---- init h state (bf16, zero-padded K=36..63) + x pad columns -----
    for (int j = tid; j < BT * 64; j += THREADS) {
        int b = j >> 6, k = j & 63;
        hs[b][k] = (k < HID) ? f2bf(h0[(size_t)(b0 + b) * HID + k])
                             : (unsigned short)0;
    }
    for (int j = tid; j < BT * 16; j += THREADS)
        xs[j >> 4][IN_DIM + (j & 15)] = 0;

    // ======================= time loop ==================================
    for (int t = 0; t < S; ++t) {
        // stage x(t): x[b, g, t*20+o] -> xs[b][g*20+o] as bf16
        for (int j = tid; j < BT * IN_DIM; j += THREADS) {
            int b = j / IN_DIM, k = j - b * IN_DIM;
            int g = k / UNITT,  o = k - g * UNITT;
            size_t gi = ((size_t)(b0 + b) * 4 + g) * (size_t)T
                      + (size_t)t * UNITT + o;
            xs[b][k] = f2bf(x[gi]);
            if (t + 4 < S) __builtin_prefetch(&x[gi + 4 * UNITT], 0, 3);
        }
        __syncthreads();   // xs(t) and hs(t) visible

        // B operands from LDS: lane reads two 16-byte chunks (k 8h..8h+7, 16+8h..16+8h+7) per tile
        BF16x16 bx[3], bh[2];
        const int koff = 8 * lhalf;
        #pragma unroll
        for (int kb = 0; kb < 3; ++kb) {
            const uint4* p = (const uint4*)&xs[lmod][32 * kb + koff];
            bx[kb].q[0] = p[0];
            bx[kb].q[1] = p[2];
        }
        #pragma unroll
        for (int kb = 0; kb < 2; ++kb) {
            const uint4* p = (const uint4*)&hs[lmod][32 * kb + koff];
            bh[kb].q[0] = p[0];
            bh[kb].q[1] = p[2];
        }

        // gates(tile) = bias + W_ih_tile @ x^T + W_hh_tile @ h^T
        F32x8 acc;
        acc.v = biasv.v;
        acc.v = __builtin_amdgcn_wmma_f32_16x16x32_bf16(false, a_ih[0].v, false, bx[0].v, (short)0, acc.v, false, false);
        acc.v = __builtin_amdgcn_wmma_f32_16x16x32_bf16(false, a_ih[1].v, false, bx[1].v, (short)0, acc.v, false, false);
        acc.v = __builtin_amdgcn_wmma_f32_16x16x32_bf16(false, a_ih[2].v, false, bx[2].v, (short)0, acc.v, false, false);
        acc.v = __builtin_amdgcn_wmma_f32_16x16x32_bf16(false, a_hh[0].v, false, bh[0].v, (short)0, acc.v, false, false);
        acc.v = __builtin_amdgcn_wmma_f32_16x16x32_bf16(false, a_hh[1].v, false, bh[1].v, (short)0, acc.v, false, false);

        // D layout -> gbuf[batch][gate]: 8 consecutive gates per lane (2x b128)
        {
            float4* dp = (float4*)&gbuf[lmod][16 * wave + 8 * lhalf];
            dp[0] = make_float4(acc.f[0], acc.f[1], acc.f[2], acc.f[3]);
            dp[1] = make_float4(acc.f[4], acc.f[5], acc.f[6], acc.f[7]);
        }
        __syncthreads();   // all 144 gates visible

        // activations + state update: 2 (b,h) pairs per lane
        #pragma unroll
        for (int q = 0; q < 2; ++q) {
            int b = bq[q], h = hq[q];
            float ig = sigm(gbuf[b][h]);
            float fg = sigm(gbuf[b][HID + h]);
            float gg = tanh_f(gbuf[b][2 * HID + h]);
            float og = sigm(gbuf[b][3 * HID + h]);
            float c  = fg * cst[q] + ig * gg;
            cst[q]   = c;
            float hn = og * tanh_f(c);
            hs[b][h] = f2bf(hn);
            if (t == S - 1) gbuf[b][h] = hn;   // final h in fp32 (cols disjoint per wave)
        }
    }
    __syncthreads();

    // ---- classifier head: relu(h @ fc1^T + b1) @ fc2^T + b2 ------------
    if (tid < BT * 16) {
        int b = tid >> 4, j = tid & 15;
        float a = fc1_b[j];
        #pragma unroll
        for (int k = 0; k < HID; ++k) a += fc1_w[j * HID + k] * gbuf[b][k];
        f1[b][j] = fmaxf(a, 0.0f);
    }
    __syncthreads();
    if (tid < BT * 3) {
        int b = tid / 3, cls = tid - b * 3;
        float a = fc2_b[cls];
        #pragma unroll
        for (int k = 0; k < 16; ++k) a += fc2_w[cls * 16 + k] * f1[b][k];
        out[(size_t)(b0 + b) * 3 + cls] = a;
    }
}

extern "C" void kernel_launch(void* const* d_in, const int* in_sizes, int n_in,
                              void* d_out, int out_size, void* d_ws, size_t ws_size,
                              hipStream_t stream) {
    const float* x    = (const float*)d_in[0];
    const float* h0   = (const float*)d_in[1];
    const float* c0   = (const float*)d_in[2];
    const float* W_ih = (const float*)d_in[3];
    const float* W_hh = (const float*)d_in[4];
    const float* b_ih = (const float*)d_in[5];
    const float* b_hh = (const float*)d_in[6];
    const float* f1w  = (const float*)d_in[7];
    const float* f1b  = (const float*)d_in[8];
    const float* f2w  = (const float*)d_in[9];
    const float* f2b  = (const float*)d_in[10];

    const int B = in_sizes[1] / HID;          // 512
    const int T = in_sizes[0] / (B * 4);      // 40960
    const int S = T / UNITT;                  // 2048 (tail trimmed like reference)

    dim3 grid(B / BT);                        // 32 blocks x 288 threads
    lstm_fused_kernel<<<grid, THREADS, 0, stream>>>(
        x, h0, c0, W_ih, W_hh, b_ih, b_hh, f1w, f1b, f2w, f2b,
        (float*)d_out, T, S);
}
